// AttentiveModule_43782896615768
// MI455X (gfx1250) — compile-verified, matrix operand order checked by reading, not run on hardware
//
#include <hip/hip_runtime.h>
#include <hip/hip_bf16.h>

// ---------------------------------------------------------------------------
// AttentiveModule for MI455X (gfx1250, wave32, WMMA).
// All GEMMs run through v_wmma_f32_16x16x32_f16 with fp32 accumulation.
// Workspace layout (requires ~208 MB):
//   [0,   64MB) : logits f32 (32MB used), later reused for z f32 (64MB)
//   [64,  80MB) : P (attention probs) f16
//   [80, 144MB) : att f32, later reused for h f32
//   [144,208MB) : y f32
// ---------------------------------------------------------------------------

typedef __attribute__((ext_vector_type(16))) _Float16 v16h;
typedef __attribute__((ext_vector_type(8)))  _Float16 v8h;
typedef __attribute__((ext_vector_type(8)))  float    v8f;

#define BM 128
#define BN 128
#define BK 32
#define LDS_PITCH 40            // halves per LDS row: 32 data + 8 pad (80B = 20 banks)
#define GEMM_THREADS 256

// k (0..31) -> permuted slot so that each lane's WMMA fragment is contiguous.
// Fragment layout (ISA 7.12.2, 16-bit A 16x32): lanes 0-15 want K 0-7,16-23;
// lanes 16-31 want K 8-15,24-31. Swap bits 3 and 4 of k:
__device__ __forceinline__ int slot_of_k(int k) {
  return (k & 7) | ((k & 8) << 1) | ((k & 16) >> 1);
}

__device__ __forceinline__ v8h cvt_f32x8(float4 a, float4 b) {
  v8h h;
  h[0] = (_Float16)a.x; h[1] = (_Float16)a.y; h[2] = (_Float16)a.z; h[3] = (_Float16)a.w;
  h[4] = (_Float16)b.x; h[5] = (_Float16)b.y; h[6] = (_Float16)b.z; h[7] = (_Float16)b.w;
  return h;
}

// Stage 16 consecutive k-elements of one row into the permuted LDS layout.
// kh is 0 or 16 (which half of the 32-wide k chunk this thread owns).
// dst = row base (&Xs[row * LDS_PITCH]).  src already points at element kh.
__device__ __forceinline__ void stage16(const float* src, _Float16* dst, int kh) {
  float4 f0 = *(const float4*)(src + 0);
  float4 f1 = *(const float4*)(src + 4);
  float4 f2 = *(const float4*)(src + 8);
  float4 f3 = *(const float4*)(src + 12);
  const int s0 = kh >> 1;                 // 0 or 8
  *(v8h*)(dst + s0)      = cvt_f32x8(f0, f1);   // k kh..kh+7   -> slots s0..s0+7
  *(v8h*)(dst + 16 + s0) = cvt_f32x8(f2, f3);   // k kh+8..+15  -> slots 16+s0..
}
__device__ __forceinline__ void stage16(const _Float16* src, _Float16* dst, int kh) {
  v8h h0 = *(const v8h*)(src);
  v8h h1 = *(const v8h*)(src + 8);
  const int s0 = kh >> 1;
  *(v8h*)(dst + s0)      = h0;
  *(v8h*)(dst + 16 + s0) = h1;
}

// ---------------------------------------------------------------------------
// Batched GEMM: C[M,N] = epilogue(A[M,K] * op(B) + bias)
//   BNT=true : B is [N,K] row-major (use B^T)        -> NT GEMM
//   BNT=false: B is [K,N] row-major (transposed in LDS) -> NN GEMM
// All of M,N must be multiples of 128 and Kd a multiple of 32 (checked by caller).
// 256 threads = 8 wave32; wave grid 2(m) x 4(n); each wave: 4x2 WMMA tiles.
// ---------------------------------------------------------------------------
template <typename TA, bool BNT, bool RELU>
__global__ __launch_bounds__(GEMM_THREADS)
void gemm_wmma_kernel(const TA* __restrict__ Ag, const float* __restrict__ Bg,
                      float* __restrict__ Cg, const float* __restrict__ bias,
                      int M, int N, int Kd, int lda, int ldb, int ldc,
                      long long strideA, long long strideB, long long strideC) {
  __shared__ _Float16 As[BM * LDS_PITCH];
  __shared__ _Float16 Bs[BN * LDS_PITCH];

  const int t  = threadIdx.x;
  const int zb = blockIdx.z;
  const TA*    A = Ag + (size_t)zb * strideA;
  const float* B = Bg + (size_t)zb * strideB;
  float*       C = Cg + (size_t)zb * strideC;

  const int m0 = blockIdx.y * BM;
  const int n0 = blockIdx.x * BN;

  // staging work split: 256 threads x 16 elems = 128 rows x 32 k
  const int arow = t >> 1;               // 0..127
  const int akh  = (t & 1) * 16;         // 0 / 16
  // NN staging split: 32 k-rows x 8 n-segments
  const int bk   = t >> 3;               // 0..31
  const int bns  = (t & 7) * 16;         // 0..112

  const int lane = t & 31;
  const int w    = t >> 5;
  const int g    = lane >> 4;            // half-wave group
  const int r    = lane & 15;            // row-within-tile / col-within-tile
  const int wm   = (w >> 2) * 64;        // wave m-base: 0 / 64
  const int wn   = (w & 3) * 32;         // wave n-base: 0/32/64/96

  v8f acc[4][2] = {};

  for (int k0 = 0; k0 < Kd; k0 += BK) {
    // ---- stage A tile ----
    stage16(A + (size_t)(m0 + arow) * lda + k0 + akh, &As[arow * LDS_PITCH], akh);
    // ---- stage B tile ----
    if (BNT) {
      stage16(B + (size_t)(n0 + arow) * ldb + k0 + akh, &Bs[arow * LDS_PITCH], akh);
    } else {
      const float* src = B + (size_t)(k0 + bk) * ldb + n0 + bns;
      float4 f0 = *(const float4*)(src + 0);
      float4 f1 = *(const float4*)(src + 4);
      float4 f2 = *(const float4*)(src + 8);
      float4 f3 = *(const float4*)(src + 12);
      const int s = slot_of_k(bk);
      _Float16* d = &Bs[(size_t)bns * LDS_PITCH + s];
      d[0 * LDS_PITCH]  = (_Float16)f0.x; d[1 * LDS_PITCH]  = (_Float16)f0.y;
      d[2 * LDS_PITCH]  = (_Float16)f0.z; d[3 * LDS_PITCH]  = (_Float16)f0.w;
      d[4 * LDS_PITCH]  = (_Float16)f1.x; d[5 * LDS_PITCH]  = (_Float16)f1.y;
      d[6 * LDS_PITCH]  = (_Float16)f1.z; d[7 * LDS_PITCH]  = (_Float16)f1.w;
      d[8 * LDS_PITCH]  = (_Float16)f2.x; d[9 * LDS_PITCH]  = (_Float16)f2.y;
      d[10 * LDS_PITCH] = (_Float16)f2.z; d[11 * LDS_PITCH] = (_Float16)f2.w;
      d[12 * LDS_PITCH] = (_Float16)f3.x; d[13 * LDS_PITCH] = (_Float16)f3.y;
      d[14 * LDS_PITCH] = (_Float16)f3.z; d[15 * LDS_PITCH] = (_Float16)f3.w;
    }
    __syncthreads();

    // ---- load fragments (two 16B LDS reads each, contiguous in permuted layout) ----
    v16h af[4], bf[2];
#pragma unroll
    for (int mt = 0; mt < 4; ++mt) {
      const v8h* p = (const v8h*)&As[(wm + mt * 16 + r) * LDS_PITCH + g * 16];
      v8h lo = p[0], hi = p[1];
      af[mt] = __builtin_shufflevector(lo, hi, 0, 1, 2, 3, 4, 5, 6, 7,
                                               8, 9, 10, 11, 12, 13, 14, 15);
    }
#pragma unroll
    for (int nt = 0; nt < 2; ++nt) {
      const v8h* p = (const v8h*)&Bs[(wn + nt * 16 + r) * LDS_PITCH + g * 16];
      v8h lo = p[0], hi = p[1];
      bf[nt] = __builtin_shufflevector(lo, hi, 0, 1, 2, 3, 4, 5, 6, 7,
                                               8, 9, 10, 11, 12, 13, 14, 15);
    }

    // ---- 8 WMMAs per k-step per wave ----
#pragma unroll
    for (int mt = 0; mt < 4; ++mt)
#pragma unroll
      for (int nt = 0; nt < 2; ++nt)
        acc[mt][nt] = __builtin_amdgcn_wmma_f32_16x16x32_f16(
            false, af[mt], false, bf[nt], (short)0, acc[mt][nt], false, false);

    __syncthreads();
  }

  // ---- epilogue: optional scalar bias + ReLU, coalesced stores ----
  const float bv = bias ? bias[0] : 0.0f;
#pragma unroll
  for (int mt = 0; mt < 4; ++mt) {
#pragma unroll
    for (int nt = 0; nt < 2; ++nt) {
      const int ncol = n0 + wn + nt * 16 + r;
#pragma unroll
      for (int rr = 0; rr < 8; ++rr) {
        const int mrow = m0 + wm + mt * 16 + g * 8 + rr;
        float v = acc[mt][nt][rr] + bv;
        if (RELU) v = fmaxf(v, 0.0f);
        C[(size_t)mrow * ldc + ncol] = v;
      }
    }
  }
}

// ---------------------------------------------------------------------------
// Masked softmax over one row of logits; writes probabilities as f16.
// grid = (QT, B), 256 threads, KT = 512 -> 2 cols/thread.
// ---------------------------------------------------------------------------
__global__ __launch_bounds__(256)
void softmax_mask_kernel(const float* __restrict__ logits, _Float16* __restrict__ P,
                         const int* __restrict__ qlen, const int* __restrict__ klen,
                         int QT, int KT, float scale) {
  __shared__ float red[256];
  const int q = blockIdx.x, b = blockIdx.y, t = threadIdx.x;
  const float MASKV = -4294967295.0f;  // -2^32 + 1
  const bool qok = q < qlen[b];
  const int  kl  = klen[b];
  const float* row = logits + ((size_t)b * QT + q) * KT;
  const int c0 = t * 2;
  float2 lv = *(const float2*)(row + c0);
  float x0 = (qok && (c0 + 0) < kl) ? lv.x * scale : MASKV;
  float x1 = (qok && (c0 + 1) < kl) ? lv.y * scale : MASKV;

  red[t] = fmaxf(x0, x1);
  __syncthreads();
  for (int s = 128; s > 0; s >>= 1) {
    if (t < s) red[t] = fmaxf(red[t], red[t + s]);
    __syncthreads();
  }
  const float rmax = red[0];
  __syncthreads();

  float e0 = __expf(x0 - rmax), e1 = __expf(x1 - rmax);
  red[t] = e0 + e1;
  __syncthreads();
  for (int s = 128; s > 0; s >>= 1) {
    if (t < s) red[t] += red[t + s];
    __syncthreads();
  }
  const float inv = 1.0f / red[0];

  _Float16* prow = P + ((size_t)b * QT + q) * KT + c0;
  prow[0] = (_Float16)(e0 * inv);
  prow[1] = (_Float16)(e1 * inv);
}

// ---------------------------------------------------------------------------
// LayerNorm over residual sum: Out[row] = LN(X[row] + R[row]) * gamma + beta.
// grid = (B*QT), 256 threads, D = 1024 -> 4 elems/thread.
// ---------------------------------------------------------------------------
__global__ __launch_bounds__(256)
void ln_residual_kernel(const float* __restrict__ X, const float* __restrict__ R,
                        const float* __restrict__ gamma, const float* __restrict__ beta,
                        float* __restrict__ Out, int D) {
  __shared__ float red[256];
  const size_t row = blockIdx.x;
  const int t = threadIdx.x;
  const float* xr = X + row * (size_t)D;
  const float* rr = R + row * (size_t)D;

  float x[4], s = 0.0f, s2 = 0.0f;
#pragma unroll
  for (int i = 0; i < 4; ++i) {
    const int c = t + 256 * i;
    const float v = xr[c] + rr[c];
    x[i] = v; s += v; s2 += v * v;
  }
  red[t] = s;  __syncthreads();
  for (int st = 128; st > 0; st >>= 1) { if (t < st) red[t] += red[t + st]; __syncthreads(); }
  const float mean = red[0] / (float)D;
  __syncthreads();
  red[t] = s2; __syncthreads();
  for (int st = 128; st > 0; st >>= 1) { if (t < st) red[t] += red[t + st]; __syncthreads(); }
  const float var  = red[0] / (float)D - mean * mean;
  const float rstd = rsqrtf(var + 1e-5f);

  float* orow = Out + row * (size_t)D;
#pragma unroll
  for (int i = 0; i < 4; ++i) {
    const int c = t + 256 * i;
    orow[c] = (x[i] - mean) * rstd * gamma[c] + beta[c];
  }
}

// ---------------------------------------------------------------------------
extern "C" void kernel_launch(void* const* d_in, const int* in_sizes, int n_in,
                              void* d_out, int out_size, void* d_ws, size_t ws_size,
                              hipStream_t stream) {
  (void)in_sizes; (void)n_in; (void)out_size; (void)ws_size;
  const int B = 32, QT = 512, KT = 512, D = 1024;
  const float scale = 0.03125f;  // 1/sqrt(1024)

  const float* Q   = (const float*)d_in[0];
  const float* K   = (const float*)d_in[1];
  const float* V   = (const float*)d_in[2];
  const int* qlen  = (const int*)d_in[3];
  const int* klen  = (const int*)d_in[4];
  const float* W1  = (const float*)d_in[5];
  const float* b1  = (const float*)d_in[6];
  const float* W2  = (const float*)d_in[7];
  const float* b2  = (const float*)d_in[8];
  const float* g1  = (const float*)d_in[9];
  const float* be1 = (const float*)d_in[10];
  const float* g2  = (const float*)d_in[11];
  const float* be2 = (const float*)d_in[12];
  float* out = (float*)d_out;

  char* ws = (char*)d_ws;
  float*    logits = (float*)(ws);                            // 32 MB (region reused by z)
  float*    zbuf   = (float*)(ws);                            // 64 MB
  _Float16* P      = (_Float16*)(ws + ((size_t)64 << 20));    // 16 MB
  float*    att    = (float*)(ws + ((size_t)80 << 20));       // 64 MB (reused by h)
  float*    hbuf   = att;
  float*    ybuf   = (float*)(ws + ((size_t)144 << 20));      // 64 MB

  dim3 blk(GEMM_THREADS);

  // 1) logits = Q @ K^T  (batched NT)
  gemm_wmma_kernel<float, true, false>
      <<<dim3(KT / BN, QT / BM, B), blk, 0, stream>>>(
          Q, K, logits, nullptr, QT, KT, D, D, D, KT,
          (long long)QT * D, (long long)KT * D, (long long)QT * KT);

  // 2) masked softmax -> P (f16)
  softmax_mask_kernel<<<dim3(QT, B), dim3(256), 0, stream>>>(
      logits, P, qlen, klen, QT, KT, scale);

  // 3) att = P @ V  (batched NN, A already f16)
  gemm_wmma_kernel<_Float16, false, false>
      <<<dim3(D / BN, QT / BM, B), blk, 0, stream>>>(
          P, V, att, nullptr, QT, D, KT, KT, D, D,
          (long long)QT * KT, (long long)KT * D, (long long)QT * D);

  // 4) y = LN(Q + att)
  ln_residual_kernel<<<dim3(B * QT), dim3(256), 0, stream>>>(Q, att, g1, be1, ybuf, D);

  // 5) h = relu(y @ W1^T + b1)   (batch folded into M)
  gemm_wmma_kernel<float, true, true>
      <<<dim3(D / BN, (B * QT) / BM, 1), blk, 0, stream>>>(
          ybuf, W1, hbuf, b1, B * QT, D, D, D, D, D, 0, 0, 0);

  // 6) z = h @ W2^T + b2
  gemm_wmma_kernel<float, true, false>
      <<<dim3(D / BN, (B * QT) / BM, 1), blk, 0, stream>>>(
          hbuf, W2, zbuf, b2, B * QT, D, D, D, D, D, 0, 0, 0);

  // 7) out = LN(y + z)
  ln_residual_kernel<<<dim3(B * QT), dim3(256), 0, stream>>>(ybuf, zbuf, g2, be2, out, D);
}
